// SelfAttnLayer_24842090840439
// MI455X (gfx1250) — matrix operand order repaired and checked by reading, not compiled, not run on hardware
//
#include <hip/hip_runtime.h>
#include <hip/hip_bf16.h>

typedef __bf16 bf16_t;
typedef __attribute__((ext_vector_type(16))) __bf16 v16bf;
typedef __attribute__((ext_vector_type(8)))  float  v8f;

struct alignas(16) U128 { unsigned int w[4]; };
union Frag { v16bf v; U128 u[2]; };

// ---------------------------------------------------------------------------
// CDNA5 async copy global -> LDS (ASYNCcnt-tracked DMA path).
// LDS operand is the 32-bit wave-relative LDS byte address (low 32 bits of a
// generic LDS pointer, per ISA 10.2 aperture mapping).
// ---------------------------------------------------------------------------
__device__ __forceinline__ unsigned lds_off32(const void* p) {
  return (unsigned)(unsigned long long)p;
}

__device__ __forceinline__ void async_copy_b128(unsigned lds_byte_off, const void* gaddr) {
  asm volatile("global_load_async_to_lds_b128 %0, %1, off"
               :: "v"(lds_byte_off), "v"(gaddr)
               : "memory");
}

__device__ __forceinline__ void wait_async0() {
  asm volatile("s_wait_asynccnt 0x0" ::: "memory");
}

// ---------------------------------------------------------------------------
// Elementwise f32 -> bf16 (vectorized x4)
// ---------------------------------------------------------------------------
__global__ void cvt_bf16_kernel(const float* __restrict__ in, bf16_t* __restrict__ out,
                                long long n) {
  long long i = ((long long)blockIdx.x * blockDim.x + threadIdx.x) * 4;
  if (i >= n) return;
  struct alignas(16) F4 { float v[4]; };
  struct alignas(8)  B4 { bf16_t v[4]; };
  F4 f = *(const F4*)(in + i);
  B4 o;
  for (int j = 0; j < 4; ++j) o.v[j] = (bf16_t)f.v[j];
  *(B4*)(out + i) = o;
}

// ---------------------------------------------------------------------------
// Transpose + convert: in[R,C] f32 -> outT[C,R] bf16  (LDS tiled 32x32)
// ---------------------------------------------------------------------------
__global__ void transpose_cvt_kernel(const float* __restrict__ in, bf16_t* __restrict__ outT,
                                     int R, int C) {
  __shared__ float tile[32][33];
  int c0 = blockIdx.x * 32, r0 = blockIdx.y * 32;
  int tx = threadIdx.x & 31, ty = threadIdx.x >> 5;   // 256 threads = 8 waves
  for (int i = ty; i < 32; i += 8)
    tile[i][tx] = in[(long long)(r0 + i) * C + (c0 + tx)];
  __syncthreads();
  for (int i = ty; i < 32; i += 8)
    outT[(long long)(c0 + i) * R + (r0 + tx)] = (bf16_t)tile[tx][i];
}

// ---------------------------------------------------------------------------
// Batched bf16 transpose of the v-slice of qkv:  vT[b][d][t] = qkv[b][t][colOff+d]
// ---------------------------------------------------------------------------
__global__ void transpose_v_kernel(const bf16_t* __restrict__ qkv, bf16_t* __restrict__ vT,
                                   int S, int D, int ld, int colOff) {
  __shared__ bf16_t tile[32][33];
  long long bin  = (long long)blockIdx.z * S * ld;
  long long bout = (long long)blockIdx.z * D * S;
  int d0 = blockIdx.x * 32, t0 = blockIdx.y * 32;
  int tx = threadIdx.x & 31, ty = threadIdx.x >> 5;
  for (int i = ty; i < 32; i += 8)
    tile[i][tx] = qkv[bin + (long long)(t0 + i) * ld + colOff + d0 + tx];
  __syncthreads();
  for (int i = ty; i < 32; i += 8)
    vT[bout + (long long)(d0 + i) * S + (t0 + tx)] = tile[tx][i];
}

// ---------------------------------------------------------------------------
// WMMA GEMM with async-LDS double buffering:
//   C[M,N] = A[M,K] * Bt[N,K]^T (+ bias[N]),  bf16 in, fp32 accumulate.
//
// Block = 128 threads (4 waves, 2Mx2N), block tile 64(M) x 128(N).
// Wave tile 32(M) x 64(N) = 2x4 WMMA tiles -> 6 LDS fragment loads feed
// 8 v_wmma_f32_16x16x32_bf16 per 32-deep K step (1.5 ds_load_b128 per wmma).
// K staged in 64-wide chunks: global_load_async_to_lds_b128 DMA-fills the
// next chunk (A: 8KB, B: 16KB) while WMMA consumes the current one
// (double buffered, ASYNCcnt + workgroup barrier). Rows padded +8 bf16.
//
// A-fragment (16-bit A 16x32): lane l<16 holds row l, K {0..7,16..23};
// lane l+16 holds row l, K {8..15,24..31} -> two 16B chunks.
// B-fragment (16-bit B 32x16): lane n<16 holds col n, K {0..7,16..23}; lane n+16 holds
// col n, K {8..15,24..31} -> two 16B chunks (rows of Bt).
// ---------------------------------------------------------------------------
#define KT 64          // K-chunk staged in LDS
#define APAD 72        // 64 + 8 bf16 row padding

template <bool OUT_BF16, bool HAS_BIAS>
__global__ void gemm_wmma_kernel(const bf16_t* __restrict__ A, long long sA, int lda,
                                 const bf16_t* __restrict__ Bt, long long sB, int ldb,
                                 void* __restrict__ Cv, long long sC, int ldc,
                                 const float* __restrict__ bias, int K) {
  __shared__ bf16_t As[2][64][APAD];    // 18 KB
  __shared__ bf16_t Bs[2][128][APAD];   // 36 KB

  const int tid  = threadIdx.x;         // 0..127
  const int lane = tid & 31;
  const int wave = tid >> 5;            // 0..3
  const int waveM = wave & 1;           // 2 waves along M (32 rows each)
  const int waveN = wave >> 1;          // 2 waves along N (64 cols each)
  const int blockM = blockIdx.y * 64;
  const int blockN = blockIdx.x * 128;
  const long long ab = (long long)blockIdx.z * sA;
  const long long bb = (long long)blockIdx.z * sB;
  const long long cb = (long long)blockIdx.z * sC;

  const int r  = lane & 15;
  const int hi = lane >> 4;

  // Issue the async DMA for one K-chunk into LDS buffer `buf`.
  auto issue_copy = [&](int buf, int kk) {
    // A tile: 64 rows x 8 chunks of 16B = 512 chunks, 4 per thread
    #pragma unroll
    for (int j = 0; j < 4; ++j) {
      int c = tid + j * 128;
      int row = c >> 3, ch = (c & 7) * 8;
      const bf16_t* g = A + ab + (long long)(blockM + row) * lda + kk + ch;
      async_copy_b128(lds_off32(&As[buf][row][ch]), g);
      if (j < 2) __builtin_prefetch(g + 2 * KT, 0, 1);  // chunk after next -> GL2
    }
    // B tile: 128 rows x 8 chunks = 1024 chunks, 8 per thread
    #pragma unroll
    for (int j = 0; j < 8; ++j) {
      int c = tid + j * 128;
      int row = c >> 3, ch = (c & 7) * 8;
      const bf16_t* g = Bt + bb + (long long)(blockN + row) * ldb + kk + ch;
      async_copy_b128(lds_off32(&Bs[buf][row][ch]), g);
    }
  };

  v8f acc[2][4];
  {
    v8f z = {};
    #pragma unroll
    for (int mi = 0; mi < 2; ++mi)
      #pragma unroll
      for (int ni = 0; ni < 4; ++ni) acc[mi][ni] = z;
  }

  issue_copy(0, 0);
  int buf = 0;
  for (int kk = 0; kk < K; kk += KT) {
    wait_async0();        // this wave's DMA into `buf` complete
    __syncthreads();      // -> all waves' DMA complete, buffer published

    if (kk + KT < K) issue_copy(buf ^ 1, kk + KT);   // overlap next chunk

    const bf16_t* Abase = &As[buf][waveM * 32 + r][0];
    const bf16_t* Bbase = &Bs[buf][waveN * 64 + r][0];

    #pragma unroll
    for (int ks = 0; ks < KT; ks += 32) {
      Frag fa[2], fb[4];
      #pragma unroll
      for (int mi = 0; mi < 2; ++mi) {
        fa[mi].u[0] = *(const U128*)(Abase + mi * 16 * APAD + ks + hi * 8);
        fa[mi].u[1] = *(const U128*)(Abase + mi * 16 * APAD + ks + hi * 8 + 16);
      }
      #pragma unroll
      for (int ni = 0; ni < 4; ++ni) {
        fb[ni].u[0] = *(const U128*)(Bbase + ni * 16 * APAD + ks + hi * 8);
        fb[ni].u[1] = *(const U128*)(Bbase + ni * 16 * APAD + ks + hi * 8 + 16);
      }
      #pragma unroll
      for (int mi = 0; mi < 2; ++mi)
        #pragma unroll
        for (int ni = 0; ni < 4; ++ni)
          acc[mi][ni] = __builtin_amdgcn_wmma_f32_16x16x32_bf16(
              false, fa[mi].v, false, fb[ni].v, (short)0, acc[mi][ni], false, false);
    }
    __syncthreads();      // everyone done reading `buf` before it is refilled
    buf ^= 1;
  }

  // C/D layout: lane<16 -> N=lane, M=vgpr idx; lane>=16 -> N=lane-16, M=8+idx
  const int tileM0 = blockM + waveM * 32;
  const int tileN0 = blockN + waveN * 64;
  const int cr = hi * 8;
  const int cc = lane & 15;

  #pragma unroll
  for (int mi = 0; mi < 2; ++mi) {
    #pragma unroll
    for (int ni = 0; ni < 4; ++ni) {
      float bv = 0.f;
      if (HAS_BIAS) bv = bias[tileN0 + ni * 16 + cc];
      if (OUT_BF16) {
        bf16_t* Cp = (bf16_t*)Cv;
        #pragma unroll
        for (int i = 0; i < 8; ++i) {
          long long ro = cb + (long long)(tileM0 + mi * 16 + cr + i) * ldc;
          Cp[ro + tileN0 + ni * 16 + cc] = (bf16_t)(acc[mi][ni][i] + bv);
        }
      } else {
        float* Cp = (float*)Cv;
        #pragma unroll
        for (int i = 0; i < 8; ++i) {
          long long ro = cb + (long long)(tileM0 + mi * 16 + cr + i) * ldc;
          Cp[ro + tileN0 + ni * 16 + cc] = acc[mi][ni][i] + bv;
        }
      }
    }
  }
}

// ---------------------------------------------------------------------------
// Stieltjes transform, Q_POW = 1: per-row 32-iter bisection for lambda s.t.
// sum 1/max(lam - x, eps) == 1. One wave32 per row (S=2048 -> 64 elems/lane,
// held in registers across all iterations).
// ---------------------------------------------------------------------------
__global__ void stieltjes_kernel(const float* __restrict__ scores,
                                 bf16_t* __restrict__ wout, int S) {
  const int wave = threadIdx.x >> 5;
  const int lane = threadIdx.x & 31;
  const long long row = (long long)blockIdx.x * 8 + wave;
  const float* rp = scores + row * S;

  float xv[64];
  float mx = -1e30f;
  #pragma unroll
  for (int i = 0; i < 64; ++i) {
    float t = rp[i * 32 + lane];
    t = fminf(fmaxf(t, -50.f), 50.f);
    xv[i] = t;
    mx = fmaxf(mx, t);
  }
  for (int o = 16; o > 0; o >>= 1) mx = fmaxf(mx, __shfl_xor(mx, o, 32));
  #pragma unroll
  for (int i = 0; i < 64; ++i) xv[i] -= mx;   // xv <= 0

  float lb = 1e-9f;
  float ub = (float)S;                         // n^(1/Q_POW), Q_POW = 1
  for (int it = 0; it < 32; ++it) {
    float mid = 0.5f * (lb + ub);
    float s = 0.f;
    #pragma unroll
    for (int i = 0; i < 64; ++i) {
      float d = fmaxf(mid - xv[i], 1e-9f);
      s += __builtin_amdgcn_rcpf(d);
    }
    for (int o = 16; o > 0; o >>= 1) s += __shfl_xor(s, o, 32);
    if (s - 1.0f > 0.f) lb = mid; else ub = mid;   // wave-uniform
  }
  float lam = 0.5f * (lb + ub);
  #pragma unroll
  for (int i = 0; i < 64; ++i) {
    float d = fmaxf(lam - xv[i], 1e-9f);
    wout[row * S + i * 32 + lane] = (bf16_t)__builtin_amdgcn_rcpf(d);
  }
}

// ---------------------------------------------------------------------------
// Final FC (N=1): out[row] = sum_d y2[row,d] * Wfc[d] + bfc. One wave per row.
// ---------------------------------------------------------------------------
__global__ void fc_kernel(const bf16_t* __restrict__ y2, const float* __restrict__ Wfc,
                          const float* __restrict__ bfc, float* __restrict__ out, int D) {
  const int wave = threadIdx.x >> 5;
  const int lane = threadIdx.x & 31;
  const long long row = (long long)blockIdx.x * 8 + wave;
  const bf16_t* rp = y2 + row * D;
  float s = 0.f;
  for (int i = lane; i < D; i += 32) s += (float)rp[i] * Wfc[i];
  for (int o = 16; o > 0; o >>= 1) s += __shfl_xor(s, o, 32);
  if (lane == 0) out[row] = s + bfc[0];
}

// ---------------------------------------------------------------------------
// Host orchestration
// ---------------------------------------------------------------------------
extern "C" void kernel_launch(void* const* d_in, const int* in_sizes, int n_in,
                              void* d_out, int out_size, void* d_ws, size_t ws_size,
                              hipStream_t stream) {
  (void)in_sizes; (void)n_in; (void)out_size; (void)ws_size;

  const float* x      = (const float*)d_in[0];
  const float* W_attn = (const float*)d_in[1];
  const float* b_attn = (const float*)d_in[2];
  const float* W_proj = (const float*)d_in[3];
  const float* b_proj = (const float*)d_in[4];
  const float* W_fc   = (const float*)d_in[5];
  const float* b_fc   = (const float*)d_in[6];
  float* out = (float*)d_out;

  const int B = 4, S = 2048, D = 2048, E3 = 6144;
  const long long MS = (long long)B * S;     // 8192 rows

  char* ws = (char*)d_ws;
  size_t off = 0;
  auto alloc = [&](size_t bytes) -> void* {
    void* p = ws + off;
    off = (off + bytes + 255) & ~(size_t)255;
    return p;
  };

  bf16_t* xb     = (bf16_t*)alloc((size_t)MS * D * 2);       //  33.5 MB
  bf16_t* WaT    = (bf16_t*)alloc((size_t)E3 * D * 2);       //  25.2 MB
  bf16_t* WpT    = (bf16_t*)alloc((size_t)D * D * 2);        //   8.4 MB
  bf16_t* qkv    = (bf16_t*)alloc((size_t)MS * E3 * 2);      // 100.7 MB
  float*  scores = (float*) alloc((size_t)B * S * S * 4);    //  67.1 MB
  bf16_t* wts    = (bf16_t*)alloc((size_t)B * S * S * 2);    //  33.5 MB
  bf16_t* vT     = (bf16_t*)alloc((size_t)B * D * S * 2);    //  33.5 MB
  bf16_t* y1     = (bf16_t*)alloc((size_t)MS * D * 2);       //  33.5 MB
  bf16_t* y2     = (bf16_t*)alloc((size_t)MS * D * 2);       //  33.5 MB

  // 1) precision conversions (weights stored transposed for contiguous B-frags)
  {
    long long n = MS * D;
    cvt_bf16_kernel<<<(unsigned)((n / 4 + 255) / 256), 256, 0, stream>>>(x, xb, n);
  }
  transpose_cvt_kernel<<<dim3(E3 / 32, D / 32), 256, 0, stream>>>(W_attn, WaT, D, E3);
  transpose_cvt_kernel<<<dim3(D / 32, D / 32),  256, 0, stream>>>(W_proj, WpT, D, D);

  // 2) qkv = x @ W_attn + b_attn          [8192 x 6144] bf16
  gemm_wmma_kernel<true, true><<<dim3(E3 / 128, MS / 64, 1), 128, 0, stream>>>(
      xb, 0, D, WaT, 0, D, (void*)qkv, 0, E3, b_attn, D);

  // 3) scores[b] = q_b @ k_b^T            [4 x 2048 x 2048] f32
  gemm_wmma_kernel<false, false><<<dim3(S / 128, S / 64, B), 128, 0, stream>>>(
      qkv, (long long)S * E3, E3,            // A = q slice (cols 0..D-1)
      qkv + D, (long long)S * E3, E3,        // Bt = k slice rows (cols D..2D-1)
      (void*)scores, (long long)S * S, S, nullptr, D);

  // 4) vT[b][d][t] = v[b][t][d]           (so attn@v B-frags are contiguous)
  transpose_v_kernel<<<dim3(D / 32, S / 32, B), 256, 0, stream>>>(qkv, vT, S, D, E3, 2 * D);

  // 5) Stieltjes bisection -> attention weights (bf16)
  stieltjes_kernel<<<(unsigned)(MS / 8), 256, 0, stream>>>(scores, wts, S);

  // 6) y1[b] = weights[b] @ v[b]          [4 x 2048 x 2048] bf16
  gemm_wmma_kernel<true, false><<<dim3(D / 128, S / 64, B), 128, 0, stream>>>(
      wts, (long long)S * S, S,
      vT, (long long)D * S, S,
      (void*)y1, (long long)S * D, D, nullptr, S);

  // 7) y2 = y1 @ W_proj + b_proj          [8192 x 2048] bf16
  gemm_wmma_kernel<true, true><<<dim3(D / 128, MS / 64, 1), 128, 0, stream>>>(
      y1, 0, D, WpT, 0, D, (void*)y2, 0, D, b_proj, D);

  // 8) out = y2 @ W_fc + b_fc             [8192] f32
  fc_kernel<<<(unsigned)(MS / 8), 256, 0, stream>>>(y2, W_fc, b_fc, out, D);
}
